// Tnn_Bin_Conv2d_57449482551314
// MI455X (gfx1250) — matrix operand order repaired and checked by reading, not compiled, not run on hardware
//
#include <hip/hip_runtime.h>
#include <hip/hip_bf16.h>
#include <stdint.h>

// ---------------------------------------------------------------------------
// Fused BN(train) -> sign binarize -> 3x3 conv (pad 1) -> bias -> ReLU
// N=32, C_in=C_out=256, H=W=56.  Implicit GEMM on v_wmma_f32_16x16x32_bf16.
// Activations channels-last (NHWC); LDS slabs staged by the Tensor Data Mover
// (TENSOR_LOAD_TO_LDS, TENSORcnt) when available, b128 copies otherwise.
// ---------------------------------------------------------------------------

typedef __attribute__((ext_vector_type(16))) __bf16          v16bf;
typedef __attribute__((ext_vector_type(16))) unsigned short  v16u;
typedef __attribute__((ext_vector_type(8)))  float           v8f;

union Frag16 { v16u u; v16bf b; };

#define Nn    32
#define Cc    256
#define Hh    56
#define Ww    56
#define HP    58
#define WP    58
#define HWs   3136          // 56*56
#define NHW   100352        // 32*3136
#define CIBLK 32            // K-chunk channels per LDS stage
#define NCIB  8             // 256/32
#define NTAP  9             // 3x3
#define SLAB_CHUNKS 928     // 4 rows * 58 cols * 4 uint4-per-pixel (64B = 32 ci)

#if __has_builtin(__builtin_amdgcn_tensor_load_to_lds) && \
    __has_builtin(__builtin_amdgcn_s_wait_tensorcnt)
#define USE_TDM 1
typedef __attribute__((ext_vector_type(4))) unsigned int v4ui;
typedef __attribute__((ext_vector_type(8))) int          v8i;
typedef __attribute__((ext_vector_type(4))) int          v4i;
#else
#define USE_TDM 0
#endif

// ---------------------------------------------------------------------------
// K1: per-channel batch mean/var  ->  fold BN into affine (s, o)
//     binarized value = sign(s*x + o)
// ---------------------------------------------------------------------------
__global__ void __launch_bounds__(256) bn_stats_kernel(
    const float* __restrict__ x, const float* __restrict__ gamma,
    const float* __restrict__ beta, float* __restrict__ scale,
    float* __restrict__ offs)
{
    __shared__ float ssum[256];
    __shared__ float ssq[256];
    const int c   = blockIdx.x;
    const int tid = threadIdx.x;
    float s = 0.f, q = 0.f;
    for (int n = 0; n < Nn; ++n) {
        const float* p = x + ((size_t)n * Cc + c) * HWs;
        for (int i = tid; i < HWs; i += 256) {
            float v = p[i];
            s += v;
            q += v * v;
        }
    }
    ssum[tid] = s;
    ssq[tid]  = q;
    __syncthreads();
    for (int off = 128; off > 0; off >>= 1) {
        if (tid < off) {
            ssum[tid] += ssum[tid + off];
            ssq[tid]  += ssq[tid + off];
        }
        __syncthreads();
    }
    if (tid == 0) {
        float mean = ssum[0] / (float)NHW;
        float var  = ssq[0] / (float)NHW - mean * mean;
        float inv  = rsqrtf(var + 1e-4f);
        float sc   = gamma[c] * inv;
        scale[c] = sc;
        offs[c]  = beta[c] - mean * sc;
    }
}

// ---------------------------------------------------------------------------
// K2a: zero the padding ring of the NHWC tensor (n, 58, 58, 256)
// ---------------------------------------------------------------------------
__global__ void __launch_bounds__(256) border_zero_kernel(
    unsigned short* __restrict__ xb, int total /* 32*58*58 */)
{
    int idx = blockIdx.x * 256 + threadIdx.x;
    if (idx >= total) return;
    int p  = idx % (HP * WP);
    int hp = p / WP, wp = p % WP;
    if (hp == 0 || hp == HP - 1 || wp == 0 || wp == WP - 1) {
        uint4 z = {0u, 0u, 0u, 0u};
        uint4* dst = (uint4*)(xb + (size_t)idx * Cc);
#pragma unroll
        for (int k = 0; k < Cc / 8; ++k) dst[k] = z;   // 256 halfs = 32 uint4
    }
}

// ---------------------------------------------------------------------------
// K2b: binarize interior to bf16 {+1,-1,0}, NCHW -> NHWC via LDS transpose.
// ---------------------------------------------------------------------------
__global__ void __launch_bounds__(256) binarize_nhwc_kernel(
    const float* __restrict__ x, const float* __restrict__ scale,
    const float* __restrict__ offs, unsigned short* __restrict__ xb)
{
    __shared__ unsigned short ldsT[Ww * CIBLK];   // [w][ci_local]
    const int b   = blockIdx.x;
    const int cb  = b & 7;              // ci block (8 of 32 channels)
    const int h   = (b >> 3) % Hh;
    const int n   = b / (8 * Hh);
    const int tid = threadIdx.x;

    for (int e = tid; e < CIBLK * Ww; e += 256) {   // coalesced read along W
        int cl = e / Ww, w0 = e % Ww;
        int c  = cb * CIBLK + cl;
        float v = scale[c] * x[(((size_t)n * Cc + c) * Hh + h) * Ww + w0] + offs[c];
        ldsT[w0 * CIBLK + cl] = (v > 0.f) ? 0x3F80u : ((v < 0.f) ? 0xBF80u : 0u);
    }
    __syncthreads();
    for (int e = tid; e < CIBLK * Ww; e += 256) {   // coalesced NHWC write
        int w0 = e / CIBLK, cl = e % CIBLK;
        xb[(((size_t)n * HP + (h + 1)) * WP + (w0 + 1)) * Cc + cb * CIBLK + cl] =
            ldsT[w0 * CIBLK + cl];
    }
}

// ---------------------------------------------------------------------------
// K3: pre-swizzle weights into WMMA A-fragment order (bf16, RNE).
// A packed in contiguous K order (same as B here):  M = lane%16,  K = j + 16*(lane>=16)
// ---------------------------------------------------------------------------
__global__ void __launch_bounds__(256) pack_weights_kernel(
    const float* __restrict__ w, unsigned short* __restrict__ Afrag, int total)
{
    int idx = blockIdx.x * 256 + threadIdx.x;
    if (idx >= total) return;
    int j    = idx & 15;
    int lane = (idx >> 4) & 31;
    int ct   = (idx >> 9) & 15;
    int t    = idx >> 13;
    int r    = t % NTAP;
    int cblk = t / NTAP;
    int kl   = j + ((lane >= 16) ? 16 : 0);
    int ci   = cblk * CIBLK + kl;
    int co   = ct * 16 + (lane & 15);
    int kh   = r / 3, kw = r % 3;
    float v  = w[(((size_t)co * Cc + ci) * 3 + kh) * 3 + kw];
    unsigned int u  = __float_as_uint(v);
    unsigned int rb = (u + 0x7FFFu + ((u >> 16) & 1u)) >> 16;   // f32->bf16 RNE
    Afrag[idx] = (unsigned short)rb;
}

// ---------------------------------------------------------------------------
// K4: implicit-GEMM binary conv, LDS double-buffered.
// Block = 8 waves = 64 co x 32 positions.  grid.x = 3136 pos-tiles * 4 co-blks.
// Slab = [row(4)][col(58)][ci(32)] bf16: the 232 pixel-rows are contiguous in
// the NHWC tensor, so the whole slab is one 2-D TDM tile (tile_dim0=32 ci,
// stride 256, tile_dim1=232).  Wave 0 drives the Tensor Data Mover for block
// i+1 while all 8 waves run the 9 WMMAs of block i from the other buffer.
// ---------------------------------------------------------------------------
__global__ void __launch_bounds__(256) bin_conv_wmma_kernel(
    const unsigned short* __restrict__ xb,
    const unsigned short* __restrict__ Afrag,
    const float* __restrict__ bias,
    float* __restrict__ out)
{
    __shared__ unsigned short lds_xb[2][4 * WP * CIBLK];   // 2 x 14848 B

    const int tid     = threadIdx.x;
    const int lane    = tid & 31;
    const int wave    = tid >> 5;
    const int coblk   = blockIdx.x & 3;          // 4 blocks of 64 co
    const int postile = blockIdx.x >> 2;         // [0, 3136)
    const int n       = (postile * 32) / HWs;    // 32 | 3136 -> never crosses images
    const int p_base  = (postile * 32) % HWs;
    const int hbase   = p_base / Ww;

    const int ct      = coblk * 4 + (wave >> 1); // 16-co tile id [0,16)
    const int nn      = lane & 15;               // C/B fragment column
    const int p_loc   = p_base + ((wave & 1) << 4) + nn;
    const int h       = p_loc / Ww;
    const int w       = p_loc % Ww;
    const int rowbase = h - hbase;               // 0 or 1
    const int khalf   = (lane >> 4) << 4;        // B layout: K = j + 16*(lane>=16)

    const int pix_start = (n * HP + hbase) * WP; // first pixel-row of the slab

#if USE_TDM
    // ---- TDM staging: one tensor_load_to_lds per K-block, issued by wave 0 --
    auto stage_issue = [&](int cblk, int buf) {
        if (wave == 0) {
            uint64_t ga = (uint64_t)(uintptr_t)xb +
                          ((uint64_t)pix_start * Cc + (unsigned)(cblk * CIBLK)) * 2u;
            unsigned lds_off = (unsigned)(uintptr_t)&lds_xb[buf][0];
            unsigned rem = (unsigned)(Nn * HP * WP - pix_start); // rows until end
            v4ui g0;
            g0.x = 1u;                                   // count=1, user descriptor
            g0.y = lds_off;                              // lds_addr (bytes)
            g0.z = (unsigned)(ga & 0xFFFFFFFFu);         // global_addr[31:0]
            g0.w = (unsigned)((ga >> 32) & 0x01FFFFFFu)  // global_addr[56:32]
                 | (2u << 30);                           // type = 2 (image)
            v8i g1;
            g1[0] = (int)(1u << 16);                     // data_size = 1 -> 2 bytes
            g1[1] = (int)((unsigned)CIBLK << 16);        // tensor_dim0[15:0] = 32
            g1[2] = (int)((rem & 0xFFFFu) << 16);        // dim0 hi=0 | tensor_dim1 lo
            g1[3] = (int)((rem >> 16) & 0xFFFFu)         // tensor_dim1 hi
                  | (int)((unsigned)CIBLK << 16);        // tile_dim0 = 32
            g1[4] = 4 * WP;                              // tile_dim1 = 232, tile_dim2=0
            g1[5] = Cc;                                  // tensor_dim0_stride = 256
            g1[6] = 0;
            g1[7] = 0;
            v4i g2 = {0, 0, 0, 0};                       // 2-D tile: groups 2/3 unused
            v4i g3 = {0, 0, 0, 0};
            v8i g4 = {0, 0, 0, 0, 0, 0, 0, 0};           // 6-arg form: extra group
            __builtin_amdgcn_tensor_load_to_lds(g0, g1, g2, g3, g4, 0);
        }
    };
    auto stage_wait = [&]() {
        if (wave == 0) __builtin_amdgcn_s_wait_tensorcnt(0);
    };
#else
    // ---- fallback staging: b128 global loads + b128 LDS stores -------------
    uint4 stage[4];
    auto load_stage = [&](int cblk) {
#pragma unroll
        for (int k = 0; k < 4; ++k) {
            int idx = tid + k * 256;
            if (idx < SLAB_CHUNKS) {
                int part = idx & 3;
                int pix  = idx >> 2;
                int row  = pix / WP;
                int wp   = pix % WP;
                int hp   = hbase + row;
                if (hp > HP - 1) hp = HP - 1;
                stage[k] = *(const uint4*)(xb +
                    (((size_t)n * HP + hp) * WP + wp) * Cc + cblk * CIBLK + part * 8);
            }
        }
    };
    auto store_stage = [&](int buf) {
#pragma unroll
        for (int k = 0; k < 4; ++k) {
            int idx = tid + k * 256;
            if (idx < SLAB_CHUNKS) {
                int part = idx & 3;
                int pix  = idx >> 2;
                *(uint4*)&lds_xb[buf][pix * CIBLK + part * 8] = stage[k];
            }
        }
    };
#endif

    // prologue: fill buffer 0
#if USE_TDM
    stage_issue(0, 0);
    stage_wait();
#else
    load_stage(0);
    store_stage(0);
#endif
    __syncthreads();

    v8f acc = {};
    for (int cblk = 0; cblk < NCIB; ++cblk) {
        if (cblk + 1 < NCIB) {
#if USE_TDM
            stage_issue(cblk + 1, (cblk + 1) & 1);       // DMA overlaps WMMA
#else
            load_stage(cblk + 1);
#endif
            __builtin_prefetch(Afrag +
                ((((size_t)(cblk + 1) * NTAP) * 16 + ct) * 32 + lane) * 16, 0, 1);
        }
        const unsigned short* slab = lds_xb[cblk & 1];
#pragma unroll
        for (int r = 0; r < NTAP; ++r) {
            const int kh = r / 3, kw = r % 3;
            // A fragment: one contiguous 32B global load per lane (pre-swizzled)
            Frag16 a;
            a.u = *(const v16u*)(Afrag +
                ((((size_t)(cblk * NTAP + r)) * 16 + ct) * 32 + lane) * 16);
            // B fragment: one contiguous 32B LDS read (16 ci at one pixel)
            Frag16 bf;
            bf.u = *(const v16u*)&slab[((rowbase + kh) * WP + (w + kw)) * CIBLK + khalf];
            acc = __builtin_amdgcn_wmma_f32_16x16x32_bf16(
                false, a.b, false, bf.b, (short)0, acc, false, false);
        }
        if (cblk + 1 < NCIB) {
#if USE_TDM
            stage_wait();                                 // TENSORcnt -> 0 (wave 0)
#else
            store_stage((cblk + 1) & 1);
#endif
        }
        __syncthreads();
    }

    // epilogue: bias + ReLU, scatter per ISA C layout (M = j + 8*(lane>=16))
#pragma unroll
    for (int j = 0; j < 8; ++j) {
        int m  = j + ((lane >> 4) << 3);
        int co = ct * 16 + m;
        float v = acc[j] + bias[co];
        out[((size_t)n * Cc + co) * HWs + p_loc] = (v > 0.f) ? v : 0.f;
    }
}

// ---------------------------------------------------------------------------
extern "C" void kernel_launch(void* const* d_in, const int* in_sizes, int n_in,
                              void* d_out, int out_size, void* d_ws, size_t ws_size,
                              hipStream_t stream)
{
    const float* x      = (const float*)d_in[0];
    const float* gamma  = (const float*)d_in[1];
    const float* beta   = (const float*)d_in[2];
    const float* conv_w = (const float*)d_in[3];
    const float* conv_b = (const float*)d_in[4];
    float* out = (float*)d_out;

    char* ws = (char*)d_ws;
    float*          scale = (float*)(ws + 0);                 // 256 f32
    float*          offs  = (float*)(ws + 1024);              // 256 f32
    unsigned short* Afrag = (unsigned short*)(ws + 4096);     // 589824 u16 = 1.13 MB
    unsigned short* xb    = (unsigned short*)(ws + (size_t)(2u << 20)); // NHWC bf16, 52.6 MB

    bn_stats_kernel<<<Cc, 256, 0, stream>>>(x, gamma, beta, scale, offs);

    const int npix = Nn * HP * WP;                            // 107,648
    border_zero_kernel<<<(npix + 255) / 256, 256, 0, stream>>>(xb, npix);

    binarize_nhwc_kernel<<<Nn * Hh * 8, 256, 0, stream>>>(x, scale, offs, xb);

    const int af_total = (NCIB * NTAP) * 16 * 32 * 16;        // 589,824
    pack_weights_kernel<<<af_total / 256, 256, 0, stream>>>(conv_w, Afrag, af_total);

    const int n_blocks = (HWs * Nn / 32) * 4;                 // 3136 pos-tiles * 4 co-blocks
    bin_conv_wmma_kernel<<<n_blocks, 256, 0, stream>>>(xb, Afrag, conv_b, out);
}
